// yolox_loss_21835613733053
// MI455X (gfx1250) — matrix operand order repaired and checked by reading, not compiled, not run on hardware
//
#include <hip/hip_runtime.h>

// ---------------------------------------------------------------------------
// YOLOX SimOTA loss for MI455X (gfx1250, wave32, WMMA).
// Pipeline: decode -> A-frag swizzle -> WMMA cls-cost + IoU/masks ->
//           per-GT wave top-k (shfl merges) -> per-anchor resolution ->
//           deterministic tree reductions.
// v2: k_cost_wmma sliced over grid.y (512 blocks, not 32) so the bandwidth-
//     dominant stage can actually occupy the MI455X's WGPs / 23.3 TB/s HBM.
// ---------------------------------------------------------------------------

typedef __attribute__((ext_vector_type(16))) _Float16 v16h;
typedef __attribute__((ext_vector_type(8)))  float    v8f;

#define B_   32
#define A_   8400      // 6400 + 1600 + 400 anchors
#define G_   50        // MAX_GT
#define NC_  80
#define KP_  96        // K padded to 3 WMMA k-steps of 32
#define NSLICE 16      // grid.y slices for the WMMA cost kernel
#define EPSF 1e-7f
#define DISF 20.0f     // 2.5 * stride0
#define BIGF 100000.0f
#define INFF 1000000000.0f
#define INSH 640.0f

__device__ __forceinline__ float sigf(float v) { return 1.0f / (1.0f + __expf(-v)); }

// ---------------------------------------------------------------------------
// Kernel 1: decode preds -> boxes, obj, sum(log(1-p)), f16 (log1mp - logp) rows
// ---------------------------------------------------------------------------
__global__ void k_decode(const float* __restrict__ p8, const float* __restrict__ p16,
                         const float* __restrict__ p32,
                         float* __restrict__ boxes, float* __restrict__ pobj,
                         float* __restrict__ s1mp, _Float16* __restrict__ Lh)
{
    int idx = blockIdx.x * blockDim.x + threadIdx.x;   // b*A_ + a
    if (idx >= B_ * A_) return;
    int b = idx / A_;
    int a = idx - b * A_;
    const float* src; int l, cell, stride;
    if (a < 6400)      { src = p8;  l = 80; cell = a;        stride = 8;  }
    else if (a < 8000) { src = p16; l = 40; cell = a - 6400; stride = 16; }
    else               { src = p32; l = 20; cell = a - 8000; stride = 32; }
    int yy = cell / l, xx = cell - yy * l;
    int cs = l * l;
    const float* base = src + (size_t)b * 85 * cs + (size_t)yy * l + xx;

    float tx = sigf(base[0]);
    float ty = sigf(base[(size_t)cs]);
    float w  = __expf(base[(size_t)2 * cs]) * (float)stride;
    float h  = __expf(base[(size_t)3 * cs]) * (float)stride;
    float x  = ((float)xx + tx) * (float)stride;
    float y  = ((float)yy + ty) * (float)stride;
    float ob = sigf(base[(size_t)4 * cs]);

    float* bx = boxes + (size_t)idx * 4;
    bx[0] = x; bx[1] = y; bx[2] = w; bx[3] = h;
    pobj[idx] = ob;

    float s = 0.0f;
    _Float16* lrow = Lh + (size_t)idx * KP_;
    for (int k = 0; k < NC_; k++) {
        float p = sigf(base[(size_t)(5 + k) * cs]);
        p = fminf(fmaxf(p, EPSF), 1.0f - EPSF);
        float lp = __logf(p), l1 = __logf(1.0f - p);
        s += l1;
        lrow[k] = (_Float16)(l1 - lp);
    }
    for (int k = NC_; k < KP_; k++) lrow[k] = (_Float16)0.0f;
    s1mp[idx] = s;
}

// ---------------------------------------------------------------------------
// Kernel 2: swizzle tcls into CDNA5 16-bit A-matrix (16x32) lane layout.
// Lane L (m = L&15, hi = L>>4), half h: v = h/2,
//   K = 16*hi + h   (contiguous, matches B loads)
// frag id = (b*4 + mtile)*3 + kstep ; per-lane 16 contiguous halves.
// ---------------------------------------------------------------------------
__global__ void k_prep_afrag(const float* __restrict__ tgt, _Float16* __restrict__ afrag)
{
    int t = blockIdx.x * blockDim.x + threadIdx.x;
    if (t >= B_ * 4 * 3 * 32) return;
    int lane = t & 31;
    int ks   = (t >> 5) % 3;
    int m    = ((t >> 5) / 3) & 3;
    int b    = (t >> 5) / 12;
    int g    = (lane & 15) + m * 16;
    int hi   = lane >> 4;
    _Float16* dst = afrag + (size_t)t * 16;
#pragma unroll
    for (int h = 0; h < 16; h++) {
        int v  = h >> 1;
        int kk = 16 * hi + h + ks * 32; (void)v;
        float val = 0.0f;
        if (g < G_ && kk < NC_)
            val = tgt[((size_t)b * G_ + g) * 85 + 5 + kk];
        dst[h] = (_Float16)val;
    }
}

// ---------------------------------------------------------------------------
// Kernel 3: WMMA cls-cost + IoU + masks.
// grid = (B_, NSLICE); each wave strides 8*NSLICE anchor tiles (wave-uniform
// loop keeps EXEC all-ones for WMMA).
// cost[g][a] = (tcls @ (log1mp-logp)^T)[g][a] - s1mp[a] + 3*iou_cost + BIG*!and
// ---------------------------------------------------------------------------
__global__ void __launch_bounds__(256)
k_cost_wmma(const _Float16* __restrict__ Lh, const _Float16* __restrict__ afrag,
            const float* __restrict__ boxes, const float* __restrict__ s1mp,
            const float* __restrict__ tgt,
            float* __restrict__ costM, float* __restrict__ iouM,
            unsigned char* __restrict__ maskM)
{
    __shared__ _Float16 saf[12 * 32 * 16];   // 12 KB: this image's A fragments
    __shared__ float    sgt[G_ * 5];         // cx,cy,w,h,flag per GT

    int b    = blockIdx.x;
    int tid  = threadIdx.x;
    int wave = tid >> 5;
    int lane = tid & 31;
    int n    = lane & 15;
    int hi   = lane >> 4;

    {   // cooperative loads into LDS
        const float4* gsrc = (const float4*)(afrag + (size_t)b * 12 * 32 * 16);
        float4* sdst = (float4*)saf;
        for (int i = tid; i < 768; i += 256) sdst[i] = gsrc[i];
        for (int i = tid; i < G_ * 5; i += 256)
            sgt[i] = tgt[((size_t)b * G_ + i / 5) * 85 + (i % 5)];
    }
    __syncthreads();

    const int NT = A_ / 16;                  // 525 anchor tiles
    for (int t = wave + 8 * blockIdx.y; t < NT; t += 8 * NSLICE) {
        int a0 = t * 16;
        int aa = a0 + n;                      // this lane's anchor column

        const _Float16* lrow = Lh + ((size_t)b * A_ + aa) * KP_ + hi * 16;
        v16h bf0 = *(const v16h*)(lrow);
        v16h bf1 = *(const v16h*)(lrow + 32);
        v16h bf2 = *(const v16h*)(lrow + 64);

        const float* bx = boxes + ((size_t)b * A_ + aa) * 4;
        float px = bx[0], py = bx[1], pw = bx[2], ph = bx[3];
        float sm = s1mp[(size_t)b * A_ + aa];
        float ptlx = px - pw * 0.5f, ptly = py - ph * 0.5f;
        float pbrx = px + pw * 0.5f, pbry = py + ph * 0.5f;

#pragma unroll
        for (int m = 0; m < 4; m++) {
            v8f acc = {};
            {
                const v16h* ap = (const v16h*)(saf + ((m * 3 + 0) * 32 + lane) * 16);
                acc = __builtin_amdgcn_wmma_f32_16x16x32_f16(false, *ap, false, bf0,
                                                             (short)0, acc, false, false);
                ap = (const v16h*)(saf + ((m * 3 + 1) * 32 + lane) * 16);
                acc = __builtin_amdgcn_wmma_f32_16x16x32_f16(false, *ap, false, bf1,
                                                             (short)0, acc, false, false);
                ap = (const v16h*)(saf + ((m * 3 + 2) * 32 + lane) * 16);
                acc = __builtin_amdgcn_wmma_f32_16x16x32_f16(false, *ap, false, bf2,
                                                             (short)0, acc, false, false);
            }
#pragma unroll
            for (int r = 0; r < 8; r++) {
                int g = m * 16 + hi * 8 + r;  // C layout: VGPR r = rows r / r+8
                if (g >= G_) continue;
                float cx = sgt[g * 5 + 0], cy = sgt[g * 5 + 1];
                float gw = sgt[g * 5 + 2], gh = sgt[g * 5 + 3];
                bool valid = (sgt[g * 5 + 4] == 1.0f);

                float Lm = fmaxf(cx - DISF, 0.0f), Rm = fminf(cx + DISF, INSH);
                float Tm = fmaxf(cy - DISF, 0.0f), Bm = fminf(cy + DISF, INSH);
                bool cm = (px >= Lm) && (px <= Rm) && (py >= Tm) && (py <= Bm);
                float hw = gw * 0.5f, hh = gh * 0.5f;
                float L2 = fmaxf(cx - hw, 0.0f), R2 = fminf(cx + hw, INSH);
                float T2 = fmaxf(cy - hh, 0.0f), B2 = fminf(cy + hh, INSH);
                bool im = (px >= L2) && (px <= R2) && (py >= T2) && (py <= B2);
                bool orm  = (cm || im) && valid;
                bool andm = (cm && im) && valid;

                float iw = fmaxf(fminf(cx + hw, pbrx) - fmaxf(cx - hw, ptlx), 0.0f);
                float ih = fmaxf(fminf(cy + hh, pbry) - fmaxf(cy - hh, ptly), 0.0f);
                float inter = iw * ih;
                float uni   = gw * gh + pw * ph - inter;
                float iou   = inter / fmaxf(uni, 1e-6f);
                float iouc  = -__logf(iou + 1e-8f);

                float cost = (acc[r] - sm) + 3.0f * iouc + (andm ? 0.0f : BIGF);
                size_t o = ((size_t)b * G_ + g) * A_ + aa;
                costM[o] = cost;
                iouM[o]  = iou;
                maskM[o] = (unsigned char)((orm ? 1 : 0) | (andm ? 2 : 0) | (valid ? 4 : 0));
            }
        }
    }
}

// ---------------------------------------------------------------------------
// Kernel 4: per-(image,GT) wave top-k.  Register top-10 per lane, shfl merge.
// ---------------------------------------------------------------------------
__global__ void k_topk(const float* __restrict__ costM, const float* __restrict__ iouM,
                       const unsigned char* __restrict__ maskM, const float* __restrict__ tgt,
                       int* __restrict__ cnt, int* __restrict__ gidx)
{
    int b = blockIdx.x / G_;
    int g = blockIdx.x - b * G_;
    int lane = threadIdx.x;
    size_t ro = ((size_t)b * G_ + g) * A_;

    float lc[10]; int li[10];      // ascending cost
    float vi[10]; int vidx[10];    // descending iou
    for (int j = 0; j < 10; j++) { lc[j] = INFF; li[j] = -1; vi[j] = -1.0f; vidx[j] = -1; }
    int andc = 0;

    for (int a = lane; a < A_; a += 32) {
        unsigned char mk = maskM[ro + a];
        andc += (mk >> 1) & 1;
        float c = INFF, u = 0.0f;
        if (mk & 1) { c = costM[ro + a]; u = iouM[ro + a]; }
        if (c < lc[9]) {
            int j = 9;
            while (j > 0 && c < lc[j - 1]) { lc[j] = lc[j - 1]; li[j] = li[j - 1]; j--; }
            lc[j] = c; li[j] = a;
        }
        if (u > vi[9]) {
            int j = 9;
            while (j > 0 && u > vi[j - 1]) { vi[j] = vi[j - 1]; vidx[j] = vidx[j - 1]; j--; }
            vi[j] = u; vidx[j] = a;
        }
    }
    for (int off = 16; off; off >>= 1) andc += __shfl_xor(andc, off, 32);

    // merge: 10 rounds of wave arg-min over each lane's list head
    int topi[10];
    int pc = 0;
    for (int j = 0; j < 10; j++) {
        float v = (pc < 10) ? lc[pc] : INFF;
        int   id = (pc < 10) ? li[pc] : 0x7fffffff;
        float bv = v; int bi = id;
        for (int off = 16; off; off >>= 1) {
            float ov = __shfl_xor(bv, off, 32);
            int   oi = __shfl_xor(bi, off, 32);
            if (ov < bv || (ov == bv && oi < bi)) { bv = ov; bi = oi; }
        }
        topi[j] = bi;
        if (pc < 10 && bi == id && bv == v) pc++;
    }
    float k10 = 0.0f;
    int pu = 0;
    for (int j = 0; j < 10; j++) {
        float v = (pu < 10) ? vi[pu] : -1.0f;
        int   id = (pu < 10) ? vidx[pu] : 0x7fffffff;
        float bv = v; int bi = id;
        for (int off = 16; off; off >>= 1) {
            float ov = __shfl_xor(bv, off, 32);
            int   oi = __shfl_xor(bi, off, 32);
            if (ov > bv || (ov == bv && oi < bi)) { bv = ov; bi = oi; }
        }
        k10 += fmaxf(bv, 0.0f);
        if (pu < 10 && bi == id && bv == v) pu++;
    }

    bool valid = (tgt[((size_t)b * G_ + g) * 85 + 4] == 1.0f);
    int dk = min(max((int)k10, 1), max(andc, 1));
    if (lane < 10) {
        int ai = topi[lane];
        bool keep = (lane < dk) && valid && (ai >= 0) && (maskM[ro + ai] & 1);
        if (keep) {
            atomicAdd(&cnt[b * A_ + ai], 1);
            gidx[b * A_ + ai] = g;   // race only when contested; overridden by argmin
        }
    }
}

// ---------------------------------------------------------------------------
// Kernel 5: per-anchor overlap resolution + losses, block tree reduce.
// ---------------------------------------------------------------------------
__global__ void __launch_bounds__(256)
k_losses(const float* __restrict__ costM, const float* __restrict__ iouM,
         const unsigned char* __restrict__ maskM, const float* __restrict__ pobj,
         const int* __restrict__ cnt, const int* __restrict__ gidx,
         float* __restrict__ partial)
{
    __shared__ float sd[3][256];
    int tid = threadIdx.x;
    int idx = blockIdx.x * 256 + tid;
    float lreg = 0.0f, lconf = 0.0f, lcls = 0.0f;
    if (idx < B_ * A_) {
        int b = idx / A_;
        int a = idx - b * A_;
        int c = cnt[idx];
        bool fg = (c > 0);
        float p = fminf(fmaxf(pobj[idx], EPSF), 1.0f - EPSF);
        lconf = fg ? -__logf(p) : -__logf(1.0f - p);
        if (fg) {
            int g;
            if (c == 1) g = gidx[idx];
            else {  // contested: argmin over cost_full column
                float best = INFF; g = 0;
                for (int gg = 0; gg < G_; gg++) {
                    size_t o = ((size_t)b * G_ + gg) * A_ + a;
                    float cv = (maskM[o] & 4) ? costM[o] : INFF;
                    if (cv < best) { best = cv; g = gg; }
                }
            }
            size_t o = ((size_t)b * G_ + g) * A_ + a;
            float iou = iouM[o];
            bool andm = maskM[o] & 2;
            // recover cls_cost from stored cost
            lcls = costM[o] - 3.0f * (-__logf(iou + 1e-8f)) - (andm ? 0.0f : BIGF);
            lreg = 1.0f - iou * iou;
        }
    }
    sd[0][tid] = lreg; sd[1][tid] = lconf; sd[2][tid] = lcls;
    __syncthreads();
    for (int o = 128; o; o >>= 1) {
        if (tid < o) {
            sd[0][tid] += sd[0][tid + o];
            sd[1][tid] += sd[1][tid + o];
            sd[2][tid] += sd[2][tid + o];
        }
        __syncthreads();
    }
    if (tid == 0) {
        partial[blockIdx.x * 3 + 0] = sd[0][0];
        partial[blockIdx.x * 3 + 1] = sd[1][0];
        partial[blockIdx.x * 3 + 2] = sd[2][0];
    }
}

// ---------------------------------------------------------------------------
// Kernel 6: final deterministic reduction + num_gts + scalar output.
// ---------------------------------------------------------------------------
__global__ void k_final(const float* __restrict__ partial, int nblk,
                        const float* __restrict__ tgt, float* __restrict__ out)
{
    __shared__ float s[4][256];
    int t = threadIdx.x;
    float r = 0.0f, cf = 0.0f, cl = 0.0f, ng = 0.0f;
    for (int i = t; i < nblk; i += 256) {
        r  += partial[i * 3 + 0];
        cf += partial[i * 3 + 1];
        cl += partial[i * 3 + 2];
    }
    for (int i = t; i < B_ * G_; i += 256)
        ng += (tgt[(size_t)i * 85 + 4] == 1.0f) ? 1.0f : 0.0f;
    s[0][t] = r; s[1][t] = cf; s[2][t] = cl; s[3][t] = ng;
    __syncthreads();
    for (int o = 128; o; o >>= 1) {
        if (t < o) {
            s[0][t] += s[0][t + o]; s[1][t] += s[1][t + o];
            s[2][t] += s[2][t + o]; s[3][t] += s[3][t + o];
        }
        __syncthreads();
    }
    if (t == 0) {
        float num = fmaxf(s[3][0], 1.0f);
        out[0] = (5.0f * s[0][0] + s[2][0] + s[1][0]) / num;
    }
}

// ---------------------------------------------------------------------------
extern "C" void kernel_launch(void* const* d_in, const int* in_sizes, int n_in,
                              void* d_out, int out_size, void* d_ws, size_t ws_size,
                              hipStream_t stream) {
    const float* p8  = (const float*)d_in[0];
    const float* p16 = (const float*)d_in[1];
    const float* p32 = (const float*)d_in[2];
    const float* tgt = (const float*)d_in[3];
    float* out = (float*)d_out;

    const size_t NA = (size_t)B_ * A_;       // 268800
    char* ws = (char*)d_ws;
    size_t off = 0;
    auto take = [&](size_t bytes) -> char* {
        char* p = ws + off;
        off = (off + bytes + 255) & ~(size_t)255;
        return p;
    };
    float*         boxes  = (float*)take(NA * 4 * sizeof(float));
    float*         pobj   = (float*)take(NA * sizeof(float));
    float*         s1mp   = (float*)take(NA * sizeof(float));
    _Float16*      Lh     = (_Float16*)take(NA * KP_ * sizeof(_Float16));
    _Float16*      afrag  = (_Float16*)take((size_t)B_ * 12 * 32 * 16 * sizeof(_Float16));
    float*         costM  = (float*)take((size_t)B_ * G_ * A_ * sizeof(float));
    float*         iouM   = (float*)take((size_t)B_ * G_ * A_ * sizeof(float));
    unsigned char* maskM  = (unsigned char*)take((size_t)B_ * G_ * A_);
    int*           cnt    = (int*)take(NA * sizeof(int));
    int*           gidx   = (int*)take(NA * sizeof(int));
    float*         part   = (float*)take(1050 * 3 * sizeof(float));

    hipMemsetAsync(cnt, 0, NA * sizeof(int), stream);

    k_decode    <<<1050, 256, 0, stream>>>(p8, p16, p32, boxes, pobj, s1mp, Lh);
    k_prep_afrag<<<48,   256, 0, stream>>>(tgt, afrag);
    dim3 cgrid(B_, NSLICE);
    k_cost_wmma <<<cgrid, 256, 0, stream>>>(Lh, afrag, boxes, s1mp, tgt, costM, iouM, maskM);
    k_topk      <<<B_ * G_, 32, 0, stream>>>(costM, iouM, maskM, tgt, cnt, gidx);
    k_losses    <<<1050, 256, 0, stream>>>(costM, iouM, maskM, pobj, cnt, gidx, part);
    k_final     <<<1,    256, 0, stream>>>(part, 1050, tgt, out);
}
